// BoundaryGreenBranch_54984171323811
// MI455X (gfx1250) — matrix operand order repaired and checked by reading, not compiled, not run on hardware
//
#include <hip/hip_runtime.h>
#include <math.h>

// CDNA5 / gfx1250, wave32. Matrix op: v_wmma_f32_16x16x32_f16.

typedef __attribute__((ext_vector_type(16))) _Float16 v16h;
typedef __attribute__((ext_vector_type(8)))  float    v8f;

#define HID 64
#define NBC 128
#define HC  64
#define WC  64
#define NC  (HC*WC)
#define NB  4

union U16h { v16h v; _Float16 a[16]; };
union U8f  { v8f  v; float    a[8];  };

// Branch-free gelu: tanh form expressed as sigmoid,
//   gelu(x) ~ x * sigmoid(1.595769122(x + 0.044715 x^3))
// Uses native v_exp_f32 + v_rcp_f32: ~6 VALU ops, no exec-mask churn.
// Error (~3e-4) is below the f16 quantization applied before the WMMA.
__device__ __forceinline__ float gelu_f(float x) {
  float x3 = x * x * x;
  float t  = 1.595769122f * x + 0.071354816575f * x3;  // 2*0.7978845608*(x + 0.044715 x^3)
  return x * __builtin_amdgcn_rcpf(1.0f + __expf(-t));
}

__device__ __forceinline__ float fast_sqrt(float x) { return __builtin_amdgcn_sqrtf(x); }

// ---------------------------------------------------------------------------
// Kernel A: per boundary point, bf = gelu(gelu(bi@e1w+e1b)@e2w+e2b),
//           A = bf @ g1w[:64]   (g1b folded into main kernel's row constants)
// One thread per (b,n); 512 total. Negligible cost.
// ---------------------------------------------------------------------------
__global__ void precompute_A(const float* __restrict__ bi,
                             const float* __restrict__ e1w, const float* __restrict__ e1b,
                             const float* __restrict__ e2w, const float* __restrict__ e2b,
                             const float* __restrict__ g1w,
                             float* __restrict__ Aout) {
  int t = blockIdx.x * blockDim.x + threadIdx.x;
  if (t >= NB * NBC) return;
  float x0 = bi[t*3+0], x1 = bi[t*3+1], x2 = bi[t*3+2];
  float h[HID];
  for (int j = 0; j < HID; ++j)
    h[j] = gelu_f(x0*e1w[j] + x1*e1w[HID + j] + x2*e1w[2*HID + j] + e1b[j]);
  float h2[HID];
  for (int j = 0; j < HID; ++j) {
    float s = e2b[j];
    for (int k = 0; k < HID; ++k) s += h[k] * e2w[k*HID + j];
    h2[j] = gelu_f(s);
  }
  for (int ko = 0; ko < HID; ++ko) {
    float s = 0.f;
    for (int j = 0; j < HID; ++j) s += h2[j] * g1w[j*HID + ko];
    Aout[t*HID + ko] = s;
  }
}

// ---------------------------------------------------------------------------
// Kernel B: fused h1 build + gelu + (h1 @ g2w) via WMMA + gelu + weighted
//           accumulation over 128 boundary points + deferred g3w reduction.
// Block = 128 threads (4 waves); each wave owns 16 coarse rows.
// grid = NB * (NC/64) = 256 blocks.
// ---------------------------------------------------------------------------
__global__ __launch_bounds__(128) void green_main(
    const float* __restrict__ bi,    // (4,128,3)
    const float* __restrict__ Aarr,  // (4,128,64) workspace
    const float* __restrict__ g1w,   // (67,64): [64],[65]=coord rows, [66]=dist row
    const float* __restrict__ g1b,   // (64)
    const float* __restrict__ g2w,   // (64,32)
    const float* __restrict__ g2b,   // (32)
    const float* __restrict__ g3w,   // (32)
    const float* __restrict__ g3bp,  // (1)
    const float* __restrict__ dsc,   // (1)
    float* __restrict__ u)           // (4,4096) workspace
{
  __shared__ float sA[NBC * HID];   // 32 KB of the 320 KB WGP LDS
  __shared__ float sBC[NBC * 2];

  const int tid  = threadIdx.x;
  const int b    = blockIdx.x >> 6;
  const int base = (blockIdx.x & 63) * 64;
  const int wave = tid >> 5;
  const int lane = tid & 31;
  const int col  = lane & 15;   // A-matrix row M / B,C,D column N
  const int half = lane >> 4;
  const int m0   = base + wave * 16;

  // cooperative preload: A rows for this batch + boundary coords
  for (int i = tid; i < NBC * HID; i += 128) sA[i] = Aarr[b * NBC * HID + i];
  for (int i = tid; i < NBC * 2;  i += 128) sBC[i] = bi[(b * NBC + (i >> 1)) * 3 + (i & 1)];
  __syncthreads();

  // K index this lane supplies at A-fragment element e (ISA 16-bit A 16x32 layout)
  int Kmap[16];
#pragma unroll
  for (int e = 0; e < 16; ++e) Kmap[e] = (e < 8) ? (e + 8 * half) : (8 + e + 8 * half);

  // per-lane A-matrix row constants: coarse@g1w_c + g1b (folded) and g1w_d
  const int   mrow = m0 + col;
  const float gxr  = -1.0f + 2.0f * (float)(mrow & 63) * (1.0f / 63.0f);
  const float gyr  = -1.0f + 2.0f * (float)(m0 >> 6)   * (1.0f / 63.0f); // iy const per tile
  float cvals[32], dvals[32];
#pragma unroll
  for (int e = 0; e < 16; ++e) {
    int K0 = Kmap[e], K1 = Kmap[e] + 32;
    cvals[e]      = gxr * g1w[64*HID + K0] + gyr * g1w[65*HID + K0] + g1b[K0];
    cvals[16 + e] = gxr * g1w[64*HID + K1] + gyr * g1w[65*HID + K1] + g1b[K1];
    dvals[e]      = g1w[66*HID + K0];
    dvals[16 + e] = g1w[66*HID + K1];
  }

  // loop-invariant B fragments: g2w (f32->f16), ISA 16-bit B 32x16 layout
  U16h B00, B10, B01, B11;
#pragma unroll
  for (int e = 0; e < 16; ++e) {
    int kr = (e < 8) ? (e + 8 * half) : (8 + e + 8 * half);
    B00.a[e] = (_Float16)g2w[(kr     ) * 32 + col];
    B01.a[e] = (_Float16)g2w[(kr     ) * 32 + 16 + col];
    B10.a[e] = (_Float16)g2w[(kr + 32) * 32 + col];
    B11.a[e] = (_Float16)g2w[(kr + 32) * 32 + 16 + col];
  }

  const float s    = fabsf(dsc[0]);
  const float g2ba = g2b[col], g2bb = g2b[16 + col];
  // coords of the 8 output rows this lane accumulates: m0 + 8*half + r
  const float gx0   = -1.0f + 2.0f * (float)((m0 & 63) + 8 * half) * (1.0f / 63.0f);
  const float gstep = 2.0f / 63.0f;

  float accA[8], accB[8], accW[8];
#pragma unroll
  for (int r = 0; r < 8; ++r) { accA[r] = 0.f; accB[r] = 0.f; accW[r] = 0.f; }

  for (int n = 0; n < NBC; ++n) {
    const float bcx = sBC[2 * n], bcy = sBC[2 * n + 1];
    const float dy  = bcy - gyr;
    const float dxo = bcx - gxr;
    const float dist = fast_sqrt(dxo * dxo + dy * dy + 1e-8f);  // for this lane's h1 row

    float dw[8];
#pragma unroll
    for (int r = 0; r < 8; ++r) {   // distance weights for the 8 C/D rows
      float ddx = bcx - (gx0 + gstep * (float)r);
      dw[r] = __expf(-s * fast_sqrt(ddx * ddx + dy * dy + 1e-8f));
    }

    // build h1 A-fragments (gelu, f32 -> f16)
    const float* An = &sA[n * HID];
    U16h a0, a1;
#pragma unroll
    for (int e = 0; e < 16; ++e) {
      int K0 = Kmap[e];
      float x0v = An[K0]      + cvals[e]      + dist * dvals[e];
      float x1v = An[K0 + 32] + cvals[16 + e] + dist * dvals[16 + e];
      a0.a[e] = (_Float16)gelu_f(x0v);
      a1.a[e] = (_Float16)gelu_f(x1v);
    }

    // h2pre(16x32) = h1(16x64) @ g2w(64x32): 2 K-steps x 2 N-halves
    v8f z = {0.f, 0.f, 0.f, 0.f, 0.f, 0.f, 0.f, 0.f};
    U8f acc0, acc1;
    acc0.v = __builtin_amdgcn_wmma_f32_16x16x32_f16(false, a0.v, false, B00.v, (short)0, z,      false, false);
    acc0.v = __builtin_amdgcn_wmma_f32_16x16x32_f16(false, a1.v, false, B10.v, (short)0, acc0.v, false, false);
    acc1.v = __builtin_amdgcn_wmma_f32_16x16x32_f16(false, a0.v, false, B01.v, (short)0, z,      false, false);
    acc1.v = __builtin_amdgcn_wmma_f32_16x16x32_f16(false, a1.v, false, B11.v, (short)0, acc1.v, false, false);

    // gelu + distance-weighted accumulation (N-reduction deferred out of loop)
#pragma unroll
    for (int r = 0; r < 8; ++r) {
      float h2a = gelu_f(acc0.a[r] + g2ba);
      float h2b = gelu_f(acc1.a[r] + g2bb);
      accA[r] += dw[r] * h2a;
      accB[r] += dw[r] * h2b;
      accW[r] += dw[r];
    }
  }

  // raw = Sum_N g3w[N]*h2acc + g3b*Sum dw ; butterfly over the 16-lane group
  const float g3a = g3w[col], g3b2 = g3w[16 + col], g3bias = g3bp[0];
#pragma unroll
  for (int r = 0; r < 8; ++r) {
    float p = accA[r] * g3a + accB[r] * g3b2;
    p += __shfl_xor(p, 1, 32);
    p += __shfl_xor(p, 2, 32);
    p += __shfl_xor(p, 4, 32);
    p += __shfl_xor(p, 8, 32);
    if (col == 0)
      u[b * NC + m0 + 8 * half + r] = (p + g3bias * accW[r]) * (1.0f / (float)NBC);
  }
}

// ---------------------------------------------------------------------------
// Kernel C: bilinear upsample 64x64 -> 256x256, align_corners
// ---------------------------------------------------------------------------
__global__ void upsample_bilinear(const float* __restrict__ u, float* __restrict__ out) {
  int idx = blockIdx.x * blockDim.x + threadIdx.x;
  if (idx >= NB * 256 * 256) return;
  int b = idx >> 16;
  int y = (idx >> 8) & 255;
  int x = idx & 255;
  float fy = (float)y * (63.0f / 255.0f);
  float fx = (float)x * (63.0f / 255.0f);
  int y0 = (int)fy, x0 = (int)fx;
  int y1 = min(y0 + 1, 63), x1 = min(x0 + 1, 63);
  float wy = fy - (float)y0, wx = fx - (float)x0;
  const float* ub = u + b * NC;
  float v00 = ub[y0 * 64 + x0], v01 = ub[y0 * 64 + x1];
  float v10 = ub[y1 * 64 + x0], v11 = ub[y1 * 64 + x1];
  float v0 = v00 * (1.f - wx) + v01 * wx;
  float v1 = v10 * (1.f - wx) + v11 * wx;
  out[idx] = v0 * (1.f - wy) + v1 * wy;
}

extern "C" void kernel_launch(void* const* d_in, const int* in_sizes, int n_in,
                              void* d_out, int out_size, void* d_ws, size_t ws_size,
                              hipStream_t stream) {
  const float* bi  = (const float*)d_in[0];
  // d_in[1] = interior_coords: unused by the reference
  const float* e1w = (const float*)d_in[2];
  const float* e1b = (const float*)d_in[3];
  const float* e2w = (const float*)d_in[4];
  const float* e2b = (const float*)d_in[5];
  const float* g1w = (const float*)d_in[6];
  const float* g1b = (const float*)d_in[7];
  const float* g2w = (const float*)d_in[8];
  const float* g2b = (const float*)d_in[9];
  const float* g3w = (const float*)d_in[10];
  const float* g3b = (const float*)d_in[11];
  const float* dsc = (const float*)d_in[12];
  (void)in_sizes; (void)n_in; (void)out_size; (void)ws_size;

  float* Aws = (float*)d_ws;              // 4*128*64 floats (128 KB)
  float* uws = Aws + NB * NBC * HID;      // 4*4096  floats (64 KB)

  precompute_A<<<2, 256, 0, stream>>>(bi, e1w, e1b, e2w, e2b, g1w, Aws);
  green_main<<<NB * 64, 128, 0, stream>>>(bi, Aws, g1w, g1b, g2w, g2b, g3w, g3b, dsc, uws);
  upsample_bilinear<<<(NB * 256 * 256) / 256, 256, 0, stream>>>(uws, (float*)d_out);
}
